// SimpleGPTModel_32366873542963
// MI455X (gfx1250) — matrix operand order repaired and checked by reading, not compiled, not run on hardware
//
#include <hip/hip_runtime.h>

// ---------------------------------------------------------------------------
// GPT forward for MI455X (gfx1250): wave32 WMMA f16->f32 GEMMs for all dense
// matmuls, VALU/LDS attention (tiny fraction of FLOPs), fused epilogues.
// ---------------------------------------------------------------------------

typedef __attribute__((ext_vector_type(16))) _Float16 v16h;
typedef __attribute__((ext_vector_type(8)))  _Float16 v8h;
typedef __attribute__((ext_vector_type(8)))  float    v8f;

#define NTOK   122880   // 2048 * 60 tokens
#define NBATCH 2048
#define TSEQ   60
#define CDIM   384
#define HHEADS 6
#define HSZ    64
#define NLAYER 6
#define FFDIM  1536
#define VOCAB  65
#define VPAD   128      // padded lm-head columns (zero-filled)
#define QKVN   1152     // fused q|k|v output width

static __device__ __forceinline__ v16h make_frag(const _Float16* p0,
                                                 const _Float16* p1) {
  v8h lo = *(const v8h*)p0;
  v8h hi = *(const v8h*)p1;
  return __builtin_shufflevector(lo, hi, 0, 1, 2, 3, 4, 5, 6, 7,
                                 8, 9, 10, 11, 12, 13, 14, 15);
}

// ---------------------------------------------------------------------------
// WMMA GEMM:  C[M,N] = A[M,K](f16) * Bt[N,K]^T(f16)  (+bias)(+residual)(relu)
// Block = 256 thr = 8 waves; block tile 128x128; wave tile 32x64
// (2 A frags x 4 B frags = 8 v_wmma_f32_16x16x32_f16 per 32-wide K step).
// A row-major [M,K]; Bt row-major [Npad,K] (i.e. B transposed).
// Stores clipped to n < nclip with leading dim ldc.
// ---------------------------------------------------------------------------
template <bool HAS_BIAS, bool RELU, bool RESID, bool OUT16>
__global__ __launch_bounds__(256) void gemm_kernel(
    const _Float16* __restrict__ A, const _Float16* __restrict__ Bt,
    const float* __restrict__ bias, const float* __restrict__ res,
    float* __restrict__ C32, _Float16* __restrict__ C16,
    int K, int nclip, int ldc) {
  const int lane  = threadIdx.x & 31;
  const int wave  = threadIdx.x >> 5;
  const int half  = lane >> 4;   // K-half select
  const int l16   = lane & 15;   // row (A) / col (B,D) within 16
  const int tileM = blockIdx.y * 128 + (wave >> 1) * 32;
  const int tileN = blockIdx.x * 128 + (wave & 1) * 64;

  v8f acc[2][4];
#pragma unroll
  for (int i = 0; i < 2; i++)
#pragma unroll
    for (int j = 0; j < 4; j++)
#pragma unroll
      for (int r = 0; r < 8; r++) acc[i][j][r] = 0.0f;

  for (int k0 = 0; k0 < K; k0 += 32) {
    v16h af[2], bf[4];
#pragma unroll
    for (int i = 0; i < 2; i++) {
      // A 16x32 layout: lanes 0-15 hold K 0-7 & 16-23; lanes 16-31 K 8-15 & 24-31
      const _Float16* p = A + (size_t)(tileM + i * 16 + l16) * K + k0 + half * 8;
      af[i] = make_frag(p, p + 16);
    }
#pragma unroll
    for (int j = 0; j < 4; j++) {
      // B 32x16 (as Bt rows): lanes 0-15 hold K 0-7 & 16-23; lanes 16-31 K 8-15 & 24-31
      const _Float16* p = Bt + (size_t)(tileN + j * 16 + l16) * K + k0 + half * 8;
      bf[j] = make_frag(p, p + 16);
    }
#pragma unroll
    for (int i = 0; i < 2; i++)
#pragma unroll
      for (int j = 0; j < 4; j++)
        acc[i][j] = __builtin_amdgcn_wmma_f32_16x16x32_f16(
            false, af[i], false, bf[j], (short)0, acc[i][j], false, false);
  }

#pragma unroll
  for (int i = 0; i < 2; i++) {
#pragma unroll
    for (int j = 0; j < 4; j++) {
      int n = tileN + j * 16 + l16;
      if (n >= nclip) continue;
      float bv = HAS_BIAS ? bias[n] : 0.0f;
#pragma unroll
      for (int r = 0; r < 8; r++) {
        // D 16x16 f32: VGPR r, lanes 0-15 -> M=r ; lanes 16-31 -> M=r+8
        int m = tileM + i * 16 + half * 8 + r;
        float v = acc[i][j][r] + bv;
        if (RESID) v += res[(size_t)m * ldc + n];
        if (RELU) v = v > 0.0f ? v : 0.0f;
        if (OUT16)
          C16[(size_t)m * ldc + n] = (_Float16)v;
        else
          C32[(size_t)m * ldc + n] = v;
      }
    }
  }
}

// --------------------------- LayerNorm (f32 -> f16) -------------------------
__global__ __launch_bounds__(256) void ln_kernel(const float* __restrict__ x,
                                                 const float* __restrict__ g,
                                                 const float* __restrict__ b,
                                                 _Float16* __restrict__ out) {
  int wave = threadIdx.x >> 5, lane = threadIdx.x & 31;
  int row = blockIdx.x * 8 + wave;  // one wave32 per row of 384
  const float* xr = x + (size_t)row * CDIM;
  float v[12];
  float s = 0.0f;
#pragma unroll
  for (int t = 0; t < 12; t++) {
    v[t] = xr[lane + t * 32];
    s += v[t];
  }
#pragma unroll
  for (int o = 16; o > 0; o >>= 1) s += __shfl_xor(s, o, 32);
  float mu = s * (1.0f / CDIM);
  float s2 = 0.0f;
#pragma unroll
  for (int t = 0; t < 12; t++) {
    float d = v[t] - mu;
    s2 += d * d;
  }
#pragma unroll
  for (int o = 16; o > 0; o >>= 1) s2 += __shfl_xor(s2, o, 32);
  float rstd = rsqrtf(s2 * (1.0f / CDIM) + 1e-5f);
  _Float16* orow = out + (size_t)row * CDIM;
#pragma unroll
  for (int t = 0; t < 12; t++) {
    int c = lane + t * 32;
    orow[c] = (_Float16)((v[t] - mu) * rstd * g[c] + b[c]);
  }
}

// ------------------------------- Embedding ----------------------------------
__global__ void embed_kernel(const int* __restrict__ ctx,
                             const float* __restrict__ tok,
                             const float* __restrict__ pos,
                             float* __restrict__ x) {
  int i = blockIdx.x * blockDim.x + threadIdx.x;
  if (i >= NTOK * CDIM) return;
  int n = i / CDIM, c = i - n * CDIM;
  int t = n % TSEQ;
  x[i] = tok[(size_t)ctx[n] * CDIM + c] + pos[t * CDIM + c];
}

// ----------------------- Attention (per batch*head) -------------------------
// qkv: [NTOK, 1152] f16 (q | k | v, head-major columns). out: [NTOK, 384] f16.
__global__ __launch_bounds__(64) void attn_kernel(
    const _Float16* __restrict__ qkv, _Float16* __restrict__ o) {
  __shared__ _Float16 ks[TSEQ][HSZ];
  __shared__ _Float16 vs[TSEQ][HSZ];
  __shared__ float sp[TSEQ][TSEQ];
  int bh = blockIdx.x;
  int b = bh / HHEADS, h = bh % HHEADS;
  const _Float16* base = qkv + (size_t)b * TSEQ * QKVN;
  int tx = threadIdx.x;
  for (int i = tx; i < TSEQ * HSZ; i += 64) {
    int t = i >> 6, d = i & 63;
    ks[t][d] = base[(size_t)t * QKVN + CDIM + h * HSZ + d];
    vs[t][d] = base[(size_t)t * QKVN + 2 * CDIM + h * HSZ + d];
  }
  __syncthreads();
  if (tx < TSEQ) {
    float q[HSZ];
    const _Float16* qp = base + (size_t)tx * QKVN + h * HSZ;
#pragma unroll
    for (int d = 0; d < HSZ; d++) q[d] = (float)qp[d];
    float mx = -1e30f;
    for (int j = 0; j <= tx; j++) {  // causal: only j <= t
      float dot = 0.0f;
#pragma unroll
      for (int d = 0; d < HSZ; d++) dot += q[d] * (float)ks[j][d];
      dot *= 0.125f;  // HS^-0.5
      sp[tx][j] = dot;
      mx = dot > mx ? dot : mx;
    }
    float sum = 0.0f;
    for (int j = 0; j <= tx; j++) {
      float e = __expf(sp[tx][j] - mx);
      sp[tx][j] = e;
      sum += e;
    }
    float inv = 1.0f / sum;
    _Float16* orow = o + (size_t)(b * TSEQ + tx) * CDIM + h * HSZ;
    for (int d = 0; d < HSZ; d++) {
      float a = 0.0f;
      for (int j = 0; j <= tx; j++) a += sp[tx][j] * (float)vs[j][d];
      orow[d] = (_Float16)(a * inv);
    }
  }
}

// ------------------------- Weight prep (f32 -> f16^T) -----------------------
// Batched transpose: src [batch,K,N] f32 -> dst [batch,N,K] f16
__global__ void transpose_f16(const float* __restrict__ src,
                              _Float16* __restrict__ dst, int batch, int K,
                              int N) {
  long per = (long)K * N;
  long total = (long)batch * per;
  long stride = (long)gridDim.x * blockDim.x;
  for (long i = blockIdx.x * (long)blockDim.x + threadIdx.x; i < total;
       i += stride) {
    long bb = i / per;
    long r = i - bb * per;
    int n = (int)(r / K);
    int k = (int)(r - (long)n * K);
    dst[i] = (_Float16)src[bb * per + (long)k * N + n];
  }
}

// wq/wk/wv [L,H,C,HS] f32 -> fused Bt [L,1152,C] f16 at column block colOff
__global__ void prep_qkv_w(const float* __restrict__ src,
                           _Float16* __restrict__ dst, int colOff) {
  const int PER = HHEADS * CDIM * HSZ;
  long total = (long)NLAYER * PER;
  long stride = (long)gridDim.x * blockDim.x;
  for (long i = blockIdx.x * (long)blockDim.x + threadIdx.x; i < total;
       i += stride) {
    int l = (int)(i / PER);
    int r = (int)(i - (long)l * PER);
    int h = r / (CDIM * HSZ);
    int r2 = r - h * (CDIM * HSZ);
    int c = r2 / HSZ;
    int d = r2 - c * HSZ;
    dst[(size_t)l * QKVN * CDIM + (size_t)(colOff + h * HSZ + d) * CDIM + c] =
        (_Float16)src[i];
  }
}

__global__ void fill_zero_f16(_Float16* __restrict__ p, int n) {
  int i = blockIdx.x * blockDim.x + threadIdx.x;
  if (i < n) p[i] = (_Float16)0.0f;
}

__global__ void prep_lm_bias(const float* __restrict__ b,
                             float* __restrict__ d) {
  int i = threadIdx.x;
  if (i < VPAD) d[i] = (i < VOCAB) ? b[i] : 0.0f;
}

// ---------------------------------------------------------------------------
extern "C" void kernel_launch(void* const* d_in, const int* in_sizes, int n_in,
                              void* d_out, int out_size, void* d_ws,
                              size_t ws_size, hipStream_t stream) {
  const int* ctx = (const int*)d_in[0];
  const float* tok = (const float*)d_in[1];
  const float* pos = (const float*)d_in[2];
  const float* wq = (const float*)d_in[3];
  const float* wk = (const float*)d_in[4];
  const float* wv = (const float*)d_in[5];
  const float* wo = (const float*)d_in[6];
  const float* bo = (const float*)d_in[7];
  const float* ln1g = (const float*)d_in[8];
  const float* ln1b = (const float*)d_in[9];
  const float* ln2g = (const float*)d_in[10];
  const float* ln2b = (const float*)d_in[11];
  const float* w1 = (const float*)d_in[12];
  const float* b1 = (const float*)d_in[13];
  const float* w2 = (const float*)d_in[14];
  const float* b2 = (const float*)d_in[15];
  const float* lnfg = (const float*)d_in[16];
  const float* lnfb = (const float*)d_in[17];
  const float* lmw = (const float*)d_in[18];
  const float* lmb = (const float*)d_in[19];
  (void)in_sizes; (void)n_in; (void)out_size;

  char* ws = (char*)d_ws;
  size_t off = 0;
  auto alloc = [&](size_t bytes) -> char* {
    char* p = ws + off;
    off += (bytes + 255) & ~(size_t)255;
    return p;
  };
  float* x        = (float*)alloc((size_t)NTOK * CDIM * 4);       // residual stream
  _Float16* hbuf  = (_Float16*)alloc((size_t)NTOK * CDIM * 2);    // ln-out / attn-out
  _Float16* big   = (_Float16*)alloc((size_t)NTOK * FFDIM * 2);   // qkv / mlp acts
  _Float16* wqkvT = (_Float16*)alloc((size_t)NLAYER * QKVN * CDIM * 2);
  _Float16* woT   = (_Float16*)alloc((size_t)NLAYER * CDIM * CDIM * 2);
  _Float16* w1T   = (_Float16*)alloc((size_t)NLAYER * FFDIM * CDIM * 2);
  _Float16* w2T   = (_Float16*)alloc((size_t)NLAYER * CDIM * FFDIM * 2);
  _Float16* lmT   = (_Float16*)alloc((size_t)VPAD * CDIM * 2);
  float* lmbPad   = (float*)alloc(VPAD * 4);
  if (ws_size < off) return;  // workspace too small; nothing safe to do

  // ---- weight prep (deterministic, every call) ----
  fill_zero_f16<<<(VPAD * CDIM + 255) / 256, 256, 0, stream>>>(lmT, VPAD * CDIM);
  prep_qkv_w<<<2048, 256, 0, stream>>>(wq, wqkvT, 0);
  prep_qkv_w<<<2048, 256, 0, stream>>>(wk, wqkvT, CDIM);
  prep_qkv_w<<<2048, 256, 0, stream>>>(wv, wqkvT, 2 * CDIM);
  transpose_f16<<<2048, 256, 0, stream>>>(wo, woT, NLAYER, CDIM, CDIM);
  transpose_f16<<<2048, 256, 0, stream>>>(w1, w1T, NLAYER, CDIM, FFDIM);
  transpose_f16<<<2048, 256, 0, stream>>>(w2, w2T, NLAYER, FFDIM, CDIM);
  transpose_f16<<<256, 256, 0, stream>>>(lmw, lmT, 1, CDIM, VOCAB);
  prep_lm_bias<<<1, VPAD, 0, stream>>>(lmb, lmbPad);

  // ---- embedding ----
  embed_kernel<<<(NTOK * CDIM + 255) / 256, 256, 0, stream>>>(ctx, tok, pos, x);

  const dim3 blk(256);
  for (int l = 0; l < NLAYER; l++) {
    // LN1 -> h
    ln_kernel<<<NTOK / 8, blk, 0, stream>>>(x, ln1g + l * CDIM, ln1b + l * CDIM,
                                            hbuf);
    // fused QKV GEMM: [NTOK,384] x [384,1152] -> big (f16)
    gemm_kernel<false, false, false, true>
        <<<dim3(QKVN / 128, NTOK / 128), blk, 0, stream>>>(
            hbuf, wqkvT + (size_t)l * QKVN * CDIM, nullptr, nullptr, nullptr,
            big, CDIM, QKVN, QKVN);
    // attention -> hbuf (f16, heads concatenated)
    attn_kernel<<<NBATCH * HHEADS, 64, 0, stream>>>(big, hbuf);
    // output proj + bias + residual -> x (f32)
    gemm_kernel<true, false, true, false>
        <<<dim3(CDIM / 128, NTOK / 128), blk, 0, stream>>>(
            hbuf, woT + (size_t)l * CDIM * CDIM, bo + l * CDIM, x, x, nullptr,
            CDIM, CDIM, CDIM);
    // LN2 -> h
    ln_kernel<<<NTOK / 8, blk, 0, stream>>>(x, ln2g + l * CDIM, ln2b + l * CDIM,
                                            hbuf);
    // MLP1 + bias + relu -> big (f16)
    gemm_kernel<true, true, false, true>
        <<<dim3(FFDIM / 128, NTOK / 128), blk, 0, stream>>>(
            hbuf, w1T + (size_t)l * FFDIM * CDIM, b1 + l * FFDIM, nullptr,
            nullptr, big, CDIM, FFDIM, FFDIM);
    // MLP2 + bias + residual -> x (f32)
    gemm_kernel<true, false, true, false>
        <<<dim3(CDIM / 128, NTOK / 128), blk, 0, stream>>>(
            big, w2T + (size_t)l * CDIM * FFDIM, b2 + l * CDIM, x, x, nullptr,
            FFDIM, CDIM, CDIM);
  }
  // final LN -> h
  ln_kernel<<<NTOK / 8, blk, 0, stream>>>(x, lnfg, lnfb, hbuf);
  // LM head: [NTOK,384] x [384,128(pad)] -> d_out f32, store clipped to 65
  gemm_kernel<true, false, false, false>
      <<<dim3(1, NTOK / 128), blk, 0, stream>>>(hbuf, lmT, lmbPad, nullptr,
                                                (float*)d_out, nullptr, CDIM,
                                                VOCAB, VOCAB);
}
